// MSDeformableAttention_7413113553233
// MI455X (gfx1250) — matrix operand order repaired and checked by reading, not compiled, not run on hardware
//
#include <hip/hip_runtime.h>

// MSDA: B=4, NQ=8192, C=256, M=8, L=3, K=4, D=32
#define B_ 4
#define NQ_ 8192
#define C_ 256
#define M_ 8
// levels: 128^2, 64^2, 32^2 ; token-row bases: lvl0 [0,65536), lvl1 [65536,81920), lvl2 [81920,86016)

typedef __bf16 bf16_t;
typedef bf16_t bf16x8  __attribute__((ext_vector_type(8)));
typedef bf16_t bf16x16 __attribute__((ext_vector_type(16)));
typedef float  f32x8   __attribute__((ext_vector_type(8)));

// ---------------- conversion kernels ----------------

__global__ void cvt_query_kernel(const float* __restrict__ q, bf16_t* __restrict__ qb, int n) {
    int i = blockIdx.x * 256 + threadIdx.x;
    if (i < n) qb[i] = (bf16_t)q[i];
}

// Build transposed bf16 weights: WsaT (288x256) = [Ws|Wa]^T, WvT (256x256), WoT (256x256)
__global__ void cvt_weights_kernel(const float* __restrict__ Ws_, const float* __restrict__ Wa_,
                                   const float* __restrict__ Wv_, const float* __restrict__ Wo_,
                                   bf16_t* __restrict__ WsaT, bf16_t* __restrict__ WvT,
                                   bf16_t* __restrict__ WoT) {
    int i = blockIdx.x * 256 + threadIdx.x;
    if (i < 288 * 256) {
        int n = i >> 8, k = i & 255;
        float v = (n < 192) ? Ws_[k * 192 + n] : Wa_[k * 96 + (n - 192)];
        WsaT[i] = (bf16_t)v;
    }
    if (i < 256 * 256) {
        int n = i >> 8, k = i & 255;
        WvT[i] = (bf16_t)Wv_[k * 256 + n];
        WoT[i] = (bf16_t)Wo_[k * 256 + n];
    }
}

// (B, C, H, W) fp32 -> (b*HWl + hw, C) bf16.
// 32x32 fp32 tile staged into LDS with GLOBAL_LOAD_ASYNC_TO_LDS_B128 (ASYNCcnt path),
// then read transposed (rows padded to 36 floats: 16B-aligned rows, low bank conflicts).
__global__ void transpose_values_kernel(const float* __restrict__ v, bf16_t* __restrict__ Xv,
                                        int HWl) {
    __shared__ float ftile[32][36];
    int t = threadIdx.x;
    int tiles_hw = HWl >> 5;
    int per_b = 8 * tiles_hw;           // 8 channel-tiles of 32
    int bid = blockIdx.x;
    int b  = bid / per_b;
    int r2 = bid % per_b;
    int ct = r2 / tiles_hw;
    int ht = r2 % tiles_hw;

    // async stage: 256 threads x 16B = full 32(ch) x 32(hw) fp32 tile
    int cl = t >> 3;                    // channel row 0..31
    int xw = (t & 7) * 4;               // float offset within row (16B granules)
    const float* gp = &v[((size_t)(b * C_ + ct * 32 + cl)) * HWl + ht * 32 + xw];
    unsigned long long ga = (unsigned long long)gp;
    unsigned lds_off = (unsigned)(unsigned long long)&ftile[cl][xw];  // flat low bits = LDS offset
    asm volatile("global_load_async_to_lds_b128 %0, %1, off"
                 :: "v"(lds_off), "v"(ga) : "memory");
    asm volatile("s_wait_asynccnt 0x0" ::: "memory");
    __syncthreads();

    // transposed write-out: consecutive lanes vary channel -> contiguous bf16 stores
#pragma unroll
    for (int i = 0; i < 4; ++i) {
        int hwl = (t >> 5) + 8 * i;
        int c2  = t & 31;
        Xv[((size_t)(b * HWl + ht * 32 + hwl)) * C_ + ct * 32 + c2] = (bf16_t)ftile[c2][hwl];
    }
}

// ---------------- WMMA bf16 GEMM: Out(MrowsxN) = A(Mrowsx256) @ BT^T + bias ----------------
// BT stored transposed (N x 256) bf16. Each wave computes a 16 x (16*TN) strip,
// reusing its A fragment TN times per k-step (A traffic / TN, higher wmma:vmem ratio).
template <int TN>
__global__ void gemm_bf16_wmma_kernel(const bf16_t* __restrict__ A,
                                      const bf16_t* __restrict__ BT,
                                      const float* __restrict__ bias0,
                                      const float* __restrict__ bias1,
                                      int bias_split,
                                      float* __restrict__ Out,
                                      int groups_n, int N) {
    int wave = blockIdx.x * 8 + (threadIdx.x >> 5);
    int lane = threadIdx.x & 31;
    int tm = wave / groups_n;
    int gn = wave % groups_n;
    int g = lane >> 4;          // half-wave group
    int r = lane & 15;

    const bf16_t* arow = A + (size_t)(tm * 16 + r) * C_;
    const bf16_t* bcol[TN];
#pragma unroll
    for (int tt = 0; tt < TN; ++tt)
        bcol[tt] = BT + (size_t)((gn * TN + tt) * 16 + r) * C_;

    f32x8 c[TN];
#pragma unroll
    for (int tt = 0; tt < TN; ++tt) c[tt] = {};

#pragma unroll
    for (int it = 0; it < 8; ++it) {        // K = 256, 32 per WMMA
        int kk = it * 32;
        // A 16x32 bf16 layout: lanes0-15 K={kk..kk+7, kk+16..kk+23}; lanes16-31 shifted +8
        bf16x8 alo = *(const bf16x8*)(arow + kk + 8 * g);
        bf16x8 ahi = *(const bf16x8*)(arow + kk + 16 + 8 * g);
        if (it + 1 < 8) __builtin_prefetch(arow + kk + 64, 0, 3);
        bf16x16 a;
#pragma unroll
        for (int e = 0; e < 8; ++e) { a[e] = alo[e]; a[e + 8] = ahi[e]; }
#pragma unroll
        for (int tt = 0; tt < TN; ++tt) {
            // B 32x16 bf16 layout: lanes0-15 K={kk..kk+7, kk+16..kk+23}; lanes16-31 shifted +8
            bf16x8 blo = *(const bf16x8*)(bcol[tt] + kk + 8 * g);
            bf16x8 bhi = *(const bf16x8*)(bcol[tt] + kk + 16 + 8 * g);
            bf16x16 b;
#pragma unroll
            for (int e = 0; e < 8; ++e) { b[e] = blo[e]; b[e + 8] = bhi[e]; }
            c[tt] = __builtin_amdgcn_wmma_f32_16x16x32_bf16(false, a, false, b,
                                                            (short)0, c[tt], false, false);
        }
    }

#pragma unroll
    for (int tt = 0; tt < TN; ++tt) {
        int col = (gn * TN + tt) * 16 + r;
        float bb = (col < bias_split) ? bias0[col] : bias1[col - bias_split];
#pragma unroll
        for (int j = 0; j < 8; ++j) {       // C/D: VGPR j -> row j + 8*g
            int row = tm * 16 + j + 8 * g;
            Out[(size_t)row * N + col] = c[tt][j] + bb;
        }
    }
}

// ---------------- softmax + bilinear sampling ----------------
__device__ __forceinline__ float fetch_tok(const float* __restrict__ tok, int iy, int ix,
                                           int H, int W, int ch) {
    bool valid = (ix >= 0) & (ix < W) & (iy >= 0) & (iy < H);
    int ixc = min(max(ix, 0), W - 1);
    int iyc = min(max(iy, 0), H - 1);
    float v = tok[((size_t)iyc * W + ixc) * C_ + ch];
    return valid ? v : 0.0f;
}

// one wave per (b, q, m); lane = channel d (D=32 == wave32)
__global__ void msda_sample_kernel(const float* __restrict__ offlog,   // (32768 x 288)
                                   const float* __restrict__ refp,     // (32768 x 2)
                                   const float* __restrict__ tokens,   // (86016 x 256)
                                   bf16_t* __restrict__ outb) {        // (32768 x 256)
    int wave = blockIdx.x * 8 + (threadIdx.x >> 5);
    int lane = threadIdx.x & 31;
    int m = wave % M_;
    int rowq = wave / M_;                 // b*NQ + q
    int b = rowq / NQ_;

    // softmax over L*K = 12 logits (uniform across lanes -> scalar-cached loads)
    const float* lp = offlog + (size_t)rowq * 288 + 192 + m * 12;
    float lg[12], wts[12];
    float mx = -1e30f;
#pragma unroll
    for (int i = 0; i < 12; ++i) { lg[i] = lp[i]; mx = fmaxf(mx, lg[i]); }
    float s = 0.0f;
#pragma unroll
    for (int i = 0; i < 12; ++i) { wts[i] = __expf(lg[i] - mx); s += wts[i]; }
    float inv = 1.0f / s;

    float rx = refp[(size_t)rowq * 2 + 0];
    float ry = refp[(size_t)rowq * 2 + 1];
    int ch = m * 32 + lane;
    float acc = 0.0f;

    const int Hs[3] = {128, 64, 32};
    const int LB[3] = {0, 65536, 81920};  // per-level token row bases

#pragma unroll
    for (int l = 0; l < 3; ++l) {
        int Hl = Hs[l], Wl = Hs[l];
        const float* tok = tokens + ((size_t)(LB[l] + b * Hl * Wl)) * C_;
#pragma unroll
        for (int k = 0; k < 4; ++k) {
            const float* op = offlog + (size_t)rowq * 288 + ((m * 3 + l) * 4 + k) * 2;
            float lx = rx + op[0] / (float)Wl;
            float ly = ry + op[1] / (float)Hl;
            // grid = loc*2-1 ; x = ((g+1)*W - 1)*0.5 = loc*W - 0.5
            float x = lx * (float)Wl - 0.5f;
            float y = ly * (float)Hl - 0.5f;
            float x0f = floorf(x), y0f = floorf(y);
            float wx1 = x - x0f, wy1 = y - y0f;
            int x0 = (int)x0f, y0 = (int)y0f;
            float aw = wts[l * 4 + k] * inv;
            float v00 = fetch_tok(tok, y0,     x0,     Hl, Wl, ch);
            float v01 = fetch_tok(tok, y0,     x0 + 1, Hl, Wl, ch);
            float v10 = fetch_tok(tok, y0 + 1, x0,     Hl, Wl, ch);
            float v11 = fetch_tok(tok, y0 + 1, x0 + 1, Hl, Wl, ch);
            acc += aw * ((1.0f - wy1) * ((1.0f - wx1) * v00 + wx1 * v01) +
                         wy1 * ((1.0f - wx1) * v10 + wx1 * v11));
        }
    }
    outb[(size_t)rowq * C_ + ch] = (bf16_t)acc;
}

// ---------------- launcher ----------------
extern "C" void kernel_launch(void* const* d_in, const int* in_sizes, int n_in,
                              void* d_out, int out_size, void* d_ws, size_t ws_size,
                              hipStream_t stream) {
    const float* query  = (const float*)d_in[0];
    const float* refp   = (const float*)d_in[1];
    const float* value0 = (const float*)d_in[2];
    const float* value1 = (const float*)d_in[3];
    const float* value2 = (const float*)d_in[4];
    const float* Wv     = (const float*)d_in[5];
    const float* bv     = (const float*)d_in[6];
    const float* Ws     = (const float*)d_in[7];
    const float* bs     = (const float*)d_in[8];
    const float* Wa     = (const float*)d_in[9];
    const float* ba     = (const float*)d_in[10];
    const float* Wo     = (const float*)d_in[11];
    const float* bo     = (const float*)d_in[12];
    float* out = (float*)d_out;

    char* ws = (char*)d_ws;
    size_t o = 0;
    auto carve = [&](size_t bytes) { size_t p = o; o = (o + bytes + 255) & ~(size_t)255; return p; };
    bf16_t* WsaT   = (bf16_t*)(ws + carve((size_t)288 * 256 * 2));
    bf16_t* WvT    = (bf16_t*)(ws + carve((size_t)256 * 256 * 2));
    bf16_t* WoT    = (bf16_t*)(ws + carve((size_t)256 * 256 * 2));
    bf16_t* qb     = (bf16_t*)(ws + carve((size_t)32768 * 256 * 2));
    bf16_t* Xv     = (bf16_t*)(ws + carve((size_t)86016 * 256 * 2));
    float*  tokens = (float*) (ws + carve((size_t)86016 * 256 * 4));
    float*  offlog = (float*) (ws + carve((size_t)32768 * 288 * 4));
    bf16_t* outb   = (bf16_t*)(ws + carve((size_t)32768 * 256 * 2));
    (void)in_sizes; (void)n_in; (void)out_size; (void)ws_size;

    // 1) precision staging
    cvt_query_kernel<<<32768, 256, 0, stream>>>(query, qb, 32768 * 256);
    cvt_weights_kernel<<<288, 256, 0, stream>>>(Ws, Wa, Wv, Wo, WsaT, WvT, WoT);

    // 2) value transpose (B,C,HW) -> (rows, C) bf16, per level (async-to-LDS tiled)
    transpose_values_kernel<<<4 * 8 * 512, 256, 0, stream>>>(value0, Xv,                       16384);
    transpose_values_kernel<<<4 * 8 * 128, 256, 0, stream>>>(value1, Xv + (size_t)65536 * C_, 4096);
    transpose_values_kernel<<<4 * 8 * 32,  256, 0, stream>>>(value2, Xv + (size_t)81920 * C_, 1024);

    // 3) offsets + logits fused GEMM: (32768x256)@(256x288), split bias [bs|ba]; TN=2 (288=9*32)
    gemm_bf16_wmma_kernel<2><<<(2048 * 9) / 8, 256, 0, stream>>>(qb, WsaT, bs, ba, 192,
                                                                 offlog, 9, 288);

    // 4) value projection GEMM: (86016x256)@(256x256) + bv; TN=4 (16x64 strip per wave)
    gemm_bf16_wmma_kernel<4><<<(5376 * 4) / 8, 256, 0, stream>>>(Xv, WvT, bv, bv, 256,
                                                                 tokens, 4, 256);

    // 5) softmax + bilinear sampling + head accumulate (wave per (b,q,m), lane=channel)
    msda_sample_kernel<<<(32768 * 8) / 8, 256, 0, stream>>>(offlog, refp, tokens, outb);

    // 6) output projection GEMM: (32768x256)@(256x256) + bo -> d_out; TN=4
    gemm_bf16_wmma_kernel<4><<<(2048 * 4) / 8, 256, 0, stream>>>(outb, WoT, bo, bo, 256,
                                                                 out, 4, 256);
}
